// RelPositionMultiHeadedAttention_48988396978803
// MI455X (gfx1250) — matrix operand order repaired and checked by reading, not compiled, not run on hardware
//
#include <hip/hip_runtime.h>

// ---------------------------------------------------------------------------
// RelPositionMultiHeadedAttention for MI455X (gfx1250), bf16 WMMA everywhere.
// B=8, T=1024, F=512, H=8, D=64.
// Round 2: K/P flash tiles now staged with global_load_async_to_lds_b128
// (ASYNCcnt path) instead of VGPR round-trips.
// ---------------------------------------------------------------------------

typedef __attribute__((ext_vector_type(16))) __bf16        v16bf;
typedef __attribute__((ext_vector_type(8)))  float         v8f;
typedef __attribute__((ext_vector_type(8)))  unsigned int  v8u;

#define DEV static __device__ __forceinline__

constexpr int B_ = 8, T_ = 1024, F_ = 512, H_ = 8, D_ = 64;

// fp32 -> bf16 round-to-nearest-even
DEV unsigned short f2bf(float f) {
  unsigned int u = __builtin_bit_cast(unsigned int, f);
  u += 0x7fffu + ((u >> 16) & 1u);
  return (unsigned short)(u >> 16);
}
DEV unsigned int pack2(float a, float b) {
  return (unsigned int)f2bf(a) | ((unsigned int)f2bf(b) << 16);
}

// WMMA 16x16x32 bf16 fragment index maps (CDNA5 ISA 7.12.2, wave32):
// A (16x32, MxK): lane gives M (lane&15); VGPR j holds K pair:
//   j<4 : K = 2j   + 8*(lane>>4)
//   j>=4: K = 16 + 2(j-4) + 8*(lane>>4)
DEV int kA(int j, int half) { return ((j & 4) << 2) + ((j & 3) << 1) + (half << 3); }
// B (32x16, KxN): lane gives N (lane&15); VGPR j holds K pair, same map as A
DEV int kB(int j, int half) { return ((j & 4) << 2) + ((j & 3) << 1) + (half << 3); }

DEV v8f wmma_bf16(const v8u& a, const v8u& b, const v8f& c) {
  return __builtin_amdgcn_wmma_f32_16x16x32_bf16(
      false, __builtin_bit_cast(v16bf, a),
      false, __builtin_bit_cast(v16bf, b),
      (short)0, c, false, false);
}

// generic pointer to __shared__ -> LDS byte offset (flat addr[31:0] per ISA 10.2)
DEV unsigned lds_off(const void* p) {
  return (unsigned)(unsigned long long)p;
}

// ---------------------------------------------------------------------------
// GEMM: out = A(MxK) @ W(KxN) + bias [+ extra column biases], bf16 WMMA.
// A is fp32 (converted in the LDS-staging path) or bf16. Output bf16 or fp32.
// Block tile 128x64x32, 256 threads = 8 waves, wave tile 32x32 (2x2 WMMA).
// M % 128 == 0, N % 64 == 0, K % 32 == 0 (holds for all uses here).
// ---------------------------------------------------------------------------
constexpr int BM = 128, BN = 64, BK = 32;
constexpr int LDA = 40, LDB = 36;  // padded LDS strides (shorts)

template <bool ABF16, bool OUTF32, bool DUAL>
__global__ __launch_bounds__(256) void gemm_proj(
    const void* __restrict__ A_, const float* __restrict__ W,
    const float* __restrict__ bias, const float* __restrict__ e0,
    const float* __restrict__ e1, void* __restrict__ out0_,
    void* __restrict__ out1_, int M, int N, int K) {
  __shared__ unsigned short sA[BM * LDA];
  __shared__ unsigned short sB[BN * LDB];  // stored transposed: [n][k]

  const int t = threadIdx.x;
  const int wave = t >> 5, lane = t & 31, half = lane >> 4, lr = lane & 15;
  const int wm = wave >> 1, wn = wave & 1;
  const int n0 = blockIdx.x * BN, m0 = blockIdx.y * BM;

  v8f acc[2][2] = {};

  for (int k0 = 0; k0 < K; k0 += BK) {
    // ---- stage A tile (128x32) ----
#pragma unroll
    for (int p = 0; p < 4; ++p) {
      const int r = p * 32 + (t >> 3), c4 = t & 7;
      if (ABF16) {
        const uint2 v = *(reinterpret_cast<const uint2*>(
                             (const unsigned short*)A_ + (size_t)(m0 + r) * K + k0) + c4);
        *reinterpret_cast<uint2*>(&sA[r * LDA + c4 * 4]) = v;
      } else {
        const float4 v = *(reinterpret_cast<const float4*>(
                              (const float*)A_ + (size_t)(m0 + r) * K + k0) + c4);
        uint2 pk;
        pk.x = pack2(v.x, v.y);
        pk.y = pack2(v.z, v.w);
        *reinterpret_cast<uint2*>(&sA[r * LDA + c4 * 4]) = pk;
      }
    }
    // ---- stage W tile transposed into sB[n][k] (64 x 32) ----
#pragma unroll
    for (int p = 0; p < 2; ++p) {
      const int lin = p * 256 + t;
      const int k = lin >> 4, n4 = (lin & 15) * 4;
      const float4 v =
          *reinterpret_cast<const float4*>(W + (size_t)(k0 + k) * N + n0 + n4);
      sB[(n4 + 0) * LDB + k] = f2bf(v.x);
      sB[(n4 + 1) * LDB + k] = f2bf(v.y);
      sB[(n4 + 2) * LDB + k] = f2bf(v.z);
      sB[(n4 + 3) * LDB + k] = f2bf(v.w);
    }
    __syncthreads();

    if (k0 + BK < K) {  // emits global_prefetch_b8
      __builtin_prefetch((const float*)W + (size_t)(k0 + BK + (t >> 4)) * N + n0, 0, 1);
      if (!ABF16)
        __builtin_prefetch((const float*)A_ + (size_t)(m0 + (t >> 3)) * K + k0 + BK, 0, 1);
    }

    // ---- fragments + WMMA ----
    v8u af[2], bf[2];
#pragma unroll
    for (int mt = 0; mt < 2; ++mt) {
      const int row = wm * 32 + mt * 16 + lr;
#pragma unroll
      for (int j = 0; j < 8; ++j)
        af[mt][j] = *reinterpret_cast<const unsigned int*>(&sA[row * LDA + kA(j, half)]);
    }
#pragma unroll
    for (int nt = 0; nt < 2; ++nt) {
      const int nn = wn * 32 + nt * 16 + lr;
#pragma unroll
      for (int j = 0; j < 8; ++j)
        bf[nt][j] = *reinterpret_cast<const unsigned int*>(&sB[nn * LDB + kB(j, half)]);
    }
#pragma unroll
    for (int mt = 0; mt < 2; ++mt)
#pragma unroll
      for (int nt = 0; nt < 2; ++nt)
        acc[mt][nt] = wmma_bf16(af[mt], bf[nt], acc[mt][nt]);
    __syncthreads();
  }

  // ---- epilogue: bias (+extra) add, store ----
#pragma unroll
  for (int mt = 0; mt < 2; ++mt)
#pragma unroll
    for (int nt = 0; nt < 2; ++nt) {
      const int col = n0 + wn * 32 + nt * 16 + lr;
      const float bb = bias[col];
      const float b0 = bb + (DUAL ? e0[col] : 0.f);
      const float b1 = DUAL ? bb + e1[col] : 0.f;
#pragma unroll
      for (int r = 0; r < 8; ++r) {
        const int row = m0 + wm * 32 + mt * 16 + r + half * 8;
        const size_t idx = (size_t)row * N + col;
        const float v = acc[mt][nt][r];
        if (OUTF32) {
          ((float*)out0_)[idx] = v + b0;
        } else {
          ((unsigned short*)out0_)[idx] = f2bf(v + b0);
          if (DUAL) ((unsigned short*)out1_)[idx] = f2bf(v + b1);
        }
      }
    }
}

// ---------------------------------------------------------------------------
// Flash attention: S = (Qu·K^T + Qv·P^T)/8, mask, online softmax, ctx = P·V.
// Grid: (B*H, T/128). 256 threads = 8 waves; each wave owns 16 query rows.
// Q fragments stay in VGPRs for the whole key loop; K/P tiles are staged via
// GLOBAL_LOAD_ASYNC_TO_LDS_B128 (ASYNCcnt), V via manual transpose.
// ---------------------------------------------------------------------------
constexpr int SK = 32;                   // keys per iteration
constexpr int LK = 72, LV = 36, LP = 40; // padded LDS strides (shorts)

__global__ __launch_bounds__(256) void flash_attn(
    const unsigned short* __restrict__ qu, const unsigned short* __restrict__ qv,
    const unsigned short* __restrict__ kp, const unsigned short* __restrict__ vp,
    const unsigned short* __restrict__ pp, const unsigned char* __restrict__ mask,
    unsigned short* __restrict__ ctxout) {
  __shared__ unsigned short sK[SK * LK];      // [s][d] row-major (= K^T B-frags)
  __shared__ unsigned short sP[SK * LK];      // [s][d]
  __shared__ unsigned short sVt[D_ * LV];     // [d][s] (V transposed for PV)
  __shared__ unsigned short sPr[8][16 * LP];  // per-wave prob tile [m][s]

  const int t = threadIdx.x;
  const int wave = t >> 5, lane = t & 31, half = lane >> 4, lr = lane & 15;
  const int bh = blockIdx.x, b = bh >> 3, h = bh & 7;
  const int t0 = blockIdx.y * 128;
  const int qrow = t0 + wave * 16 + lr;

  // Q_u / Q_v A-fragments (2 k-steps of 32 over D=64), resident in VGPRs.
  v8u quf[2], qvf[2];
  {
    const unsigned int* ru =
        reinterpret_cast<const unsigned int*>(qu + (size_t)(b * T_ + qrow) * F_ + h * D_);
    const unsigned int* rv =
        reinterpret_cast<const unsigned int*>(qv + (size_t)(b * T_ + qrow) * F_ + h * D_);
#pragma unroll
    for (int kk = 0; kk < 2; ++kk)
#pragma unroll
      for (int j = 0; j < 8; ++j) {
        const int off = (kk * 32 + kA(j, half)) >> 1;
        quf[kk][j] = ru[off];
        qvf[kk][j] = rv[off];
      }
  }

  v8f ctx[4] = {};
  float mx[8], sm[8];
#pragma unroll
  for (int r = 0; r < 8; ++r) { mx[r] = -__builtin_inff(); sm[r] = 0.f; }

  for (int s0 = 0; s0 < T_; s0 += SK) {
    // ---- stage K, P via async global->LDS (b128: 16B/lane, 512B/wave) ----
    {
      const int s = t >> 3, d8 = (t & 7) * 8;
      const unsigned short* gk = kp + (size_t)(b * T_ + s0 + s) * F_ + h * D_ + d8;
      const unsigned short* gp = pp + (size_t)(s0 + s) * F_ + h * D_ + d8;
      const unsigned ldsK = lds_off(&sK[s * LK + d8]);
      const unsigned ldsP = lds_off(&sP[s * LK + d8]);
      asm volatile("global_load_async_to_lds_b128 %0, %1, off"
                   :: "v"(ldsK), "v"((unsigned long long)gk) : "memory");
      asm volatile("global_load_async_to_lds_b128 %0, %1, off"
                   :: "v"(ldsP), "v"((unsigned long long)gp) : "memory");
      // V tile: needs transpose -> manual scatter into sVt[d][s]
      union { uint4 u; unsigned short us[8]; } vv;
      vv.u = *reinterpret_cast<const uint4*>(vp + (size_t)(b * T_ + s0 + s) * F_ + h * D_ + d8);
#pragma unroll
      for (int e = 0; e < 8; ++e) sVt[(d8 + e) * LV + s] = vv.us[e];
      asm volatile("s_wait_asynccnt 0x0" ::: "memory");
    }
    __syncthreads();

    // ---- scores: two 16x16 s-subtiles, AC + BD fused ----
    v8f sc[2] = {};
#pragma unroll
    for (int st = 0; st < 2; ++st) {
      const int srow = st * 16 + lr;
#pragma unroll
      for (int kk = 0; kk < 2; ++kk) {
        v8u bk, bp;
#pragma unroll
        for (int j = 0; j < 8; ++j) {
          const int d = kk * 32 + kB(j, half);
          bk[j] = *reinterpret_cast<const unsigned int*>(&sK[srow * LK + d]);
          bp[j] = *reinterpret_cast<const unsigned int*>(&sP[srow * LK + d]);
        }
        sc[st] = wmma_bf16(quf[kk], bk, sc[st]);
        sc[st] = wmma_bf16(qvf[kk], bp, sc[st]);
      }
    }

    // ---- scale + mask (per lane column) ----
#pragma unroll
    for (int st = 0; st < 2; ++st) {
      const int scol = s0 + st * 16 + lr;
      const bool ok = mask[b * T_ + scol] != 0;
#pragma unroll
      for (int r = 0; r < 8; ++r) {
        const float v = sc[st][r] * 0.125f;
        sc[st][r] = ok ? v : -1e9f;
      }
    }

    // ---- online softmax (row stats replicated across 16-lane groups) ----
    float p0[8], p1[8];
#pragma unroll
    for (int r = 0; r < 8; ++r) {
      float tm = fmaxf(sc[0][r], sc[1][r]);
      tm = fmaxf(tm, __shfl_xor(tm, 1));
      tm = fmaxf(tm, __shfl_xor(tm, 2));
      tm = fmaxf(tm, __shfl_xor(tm, 4));
      tm = fmaxf(tm, __shfl_xor(tm, 8));
      const float nm = fmaxf(mx[r], tm);
      const float alpha = (mx[r] == -__builtin_inff()) ? 0.f : __expf(mx[r] - nm);
      mx[r] = nm;
      const float a = __expf(sc[0][r] - nm);
      const float c = __expf(sc[1][r] - nm);
      p0[r] = a;
      p1[r] = c;
      float rs = a + c;
      rs += __shfl_xor(rs, 1);
      rs += __shfl_xor(rs, 2);
      rs += __shfl_xor(rs, 4);
      rs += __shfl_xor(rs, 8);
      sm[r] = sm[r] * alpha + rs;
#pragma unroll
      for (int nt = 0; nt < 4; ++nt) ctx[nt][r] *= alpha;
    }

    // ---- probs: C-layout -> A-fragment relayout via per-wave LDS ----
#pragma unroll
    for (int r = 0; r < 8; ++r) {
      sPr[wave][(r + half * 8) * LP + lr] = f2bf(p0[r]);
      sPr[wave][(r + half * 8) * LP + 16 + lr] = f2bf(p1[r]);
    }
    // same-wave LDS ops are in-order (DS in-order per wave) -> safe RAW
    v8u ap;
#pragma unroll
    for (int j = 0; j < 8; ++j)
      ap[j] = *reinterpret_cast<const unsigned int*>(&sPr[wave][lr * LP + kA(j, half)]);

    // ---- ctx += P(16x32) @ V(32x64): 4 WMMA over d-subtiles ----
#pragma unroll
    for (int nt = 0; nt < 4; ++nt) {
      v8u bv;
#pragma unroll
      for (int j = 0; j < 8; ++j)
        bv[j] = *reinterpret_cast<const unsigned int*>(&sVt[(nt * 16 + lr) * LV + kB(j, half)]);
      ctx[nt] = wmma_bf16(ap, bv, ctx[nt]);
    }
    __syncthreads();
  }

  // ---- normalize, write ctx (bf16) in (B,T,F) layout ----
#pragma unroll
  for (int nt = 0; nt < 4; ++nt)
#pragma unroll
    for (int r = 0; r < 8; ++r) {
      const int row = t0 + wave * 16 + r + half * 8;
      const int col = h * D_ + nt * 16 + lr;
      ctxout[(size_t)(b * T_ + row) * F_ + col] = f2bf(ctx[nt][r] / sm[r]);
    }
}

// ---------------------------------------------------------------------------
extern "C" void kernel_launch(void* const* d_in, const int* in_sizes, int n_in,
                              void* d_out, int out_size, void* d_ws, size_t ws_size,
                              hipStream_t stream) {
  const float* query = (const float*)d_in[0];
  const float* key   = (const float*)d_in[1];
  const float* value = (const float*)d_in[2];
  const float* pos   = (const float*)d_in[3];
  const unsigned char* mask = (const unsigned char*)d_in[4];
  const float* Wq = (const float*)d_in[5];
  const float* bq = (const float*)d_in[6];
  const float* Wk = (const float*)d_in[7];
  const float* bk = (const float*)d_in[8];
  const float* Wv = (const float*)d_in[9];
  const float* bv = (const float*)d_in[10];
  const float* Wp = (const float*)d_in[11];
  const float* bp = (const float*)d_in[12];
  const float* pbu = (const float*)d_in[13];
  const float* pbv = (const float*)d_in[14];
  const float* Wo = (const float*)d_in[15];
  const float* bo = (const float*)d_in[16];

  // bf16 workspace buffers
  unsigned short* ws = (unsigned short*)d_ws;
  const size_t MT = (size_t)B_ * T_ * F_;  // 4M elements
  unsigned short* quB = ws;                // q + bias + pos_bias_u
  unsigned short* qvB = quB + MT;          // q + bias + pos_bias_v
  unsigned short* kpB = qvB + MT;
  unsigned short* vpB = kpB + MT;
  unsigned short* ppB = vpB + MT;          // T_*F_
  unsigned short* ctxB = ppB + (size_t)T_ * F_;

  const dim3 blk(256);
  const dim3 gProj(F_ / BN, (B_ * T_) / BM);  // (8, 64)
  const dim3 gPos(F_ / BN, T_ / BM);          // (8, 8)
  const dim3 gFA(B_ * H_, T_ / 128);          // (64, 8)

  gemm_proj<false, false, true><<<gProj, blk, 0, stream>>>(
      query, Wq, bq, pbu, pbv, quB, qvB, B_ * T_, F_, F_);
  gemm_proj<false, false, false><<<gProj, blk, 0, stream>>>(
      key, Wk, bk, nullptr, nullptr, kpB, nullptr, B_ * T_, F_, F_);
  gemm_proj<false, false, false><<<gProj, blk, 0, stream>>>(
      value, Wv, bv, nullptr, nullptr, vpB, nullptr, B_ * T_, F_, F_);
  gemm_proj<false, false, false><<<gPos, blk, 0, stream>>>(
      pos, Wp, bp, nullptr, nullptr, ppB, nullptr, T_, F_, F_);
  flash_attn<<<gFA, blk, 0, stream>>>(quB, qvB, kpB, vpB, ppB, mask, ctxB);
  gemm_proj<true, true, false><<<gProj, blk, 0, stream>>>(
      ctxB, Wo, bo, nullptr, nullptr, d_out, nullptr, B_ * T_, F_, F_);
}
